// MHSAModule_8546984919345
// MI455X (gfx1250) — matrix operand order repaired and checked by reading, not compiled, not run on hardware
//
#include <hip/hip_runtime.h>

// ---------------- CDNA5 WMMA helpers (wave32, 16x16x32 f16 -> f32) ----------
typedef _Float16 v16h __attribute__((ext_vector_type(16)));
typedef _Float16 h8   __attribute__((ext_vector_type(8)));
typedef float    v8f  __attribute__((ext_vector_type(8)));
typedef unsigned int ui4 __attribute__((ext_vector_type(4)));
typedef int          i8v __attribute__((ext_vector_type(8)));
typedef int          i4v __attribute__((ext_vector_type(4)));

#define NB   16
#define NS   1025
#define NSP  1056          // S padded to multiple of 32 (PV K-loop, TDM K rows)
#define ND   768
#define NH   12
#define NHD  64
#define NM   (NB * NS)     // 16400 rows

static __device__ __forceinline__ v16h mk16(h8 lo, h8 hi) {
  return __builtin_shufflevector(lo, hi, 0,1,2,3,4,5,6,7,8,9,10,11,12,13,14,15);
}
static __device__ __forceinline__ v8f wmma16(v16h a, v16h b, v8f c) {
  return __builtin_amdgcn_wmma_f32_16x16x32_f16(false, a, false, b, (short)0, c,
                                                false, false);
}

// Tensor Data Mover: load a 16-row x 64-col f16 tile (row stride 64 elems)
// from global memory into LDS at byte offset lds_off. D# per ISA 08 §8.3/8.4.
static __device__ __forceinline__ void tdm_load_k16x64(const _Float16* gsrc,
                                                       unsigned lds_off) {
  unsigned long long ga = (unsigned long long)(size_t)gsrc;
  ui4 g0 = { 1u,                                   // count=1 (valid descriptor)
             lds_off,                              // lds_addr [63:32]
             (unsigned)ga,                         // global_addr low
             (unsigned)(ga >> 32) | (2u << 30) };  // addr[56:32] | type=2
  i8v g1 = { (int)0x00010000u,        // workgroup_mask=0, data_size=2B
             (int)(64u << 16),        // tensor_dim0 = 64 (bits 63:48)
             (int)(1056u << 16),      // tensor_dim1 = 1056 (bits 95:80)
             (int)(64u << 16),        // tile_dim0 = 64 (bits 127:112)
             16,                      // tile_dim1 = 16 (bits 143:128)
             64,                      // tensor_dim0_stride = 64 (bits 191:160)
             0, 0 };
  i4v z4 = { 0, 0, 0, 0 };
#if __clang_major__ >= 23
  i8v z8 = { 0, 0, 0, 0, 0, 0, 0, 0 };
  __builtin_amdgcn_tensor_load_to_lds(g0, g1, z4, z4, z8, 0);
#else
  __builtin_amdgcn_tensor_load_to_lds(g0, g1, z4, z4, 0);
#endif
}

// ---------------- stage 1: fp32 -> f16 conversion ---------------------------
__global__ void __launch_bounds__(256)
cvt_f32_f16(const float* __restrict__ s, _Float16* __restrict__ d, int n) {
  int i = blockIdx.x * blockDim.x + threadIdx.x;
  int stride = gridDim.x * blockDim.x;
  for (; i < n; i += stride) d[i] = (_Float16)s[i];
}

// ---------------- stage 2: fused QKV projection GEMM ------------------------
// grid: (513 m-tiles of 32 rows, 12 heads, 3 proj), 1 wave. Software-pipelined:
// next K-step's A/B fragments load while current 8 WMMAs issue.
__global__ void __launch_bounds__(32)
qkv_gemm(const _Float16* __restrict__ xh, const _Float16* __restrict__ wh,
         const float* __restrict__ bq, const float* __restrict__ bk,
         const float* __restrict__ bv,
         _Float16* __restrict__ qo, _Float16* __restrict__ ko,
         _Float16* __restrict__ vo) {
  const int proj = blockIdx.z;
  const int h    = blockIdx.y;
  const int m0   = blockIdx.x * 32;
  const int lane = threadIdx.x;
  const int half = lane >> 4;
  const int lm   = lane & 15;

  const _Float16* W    = wh + (size_t)proj * ND * ND;
  const float*    bias = (proj == 0) ? bq : (proj == 1) ? bk : bv;

  int ra0 = m0 + lm;      if (ra0 > NM - 1) ra0 = NM - 1;
  int ra1 = m0 + 16 + lm; if (ra1 > NM - 1) ra1 = NM - 1;
  const _Float16* arow0 = xh + (size_t)ra0 * ND;
  const _Float16* arow1 = xh + (size_t)ra1 * ND;
  const _Float16* bp[4];
#pragma unroll
  for (int t = 0; t < 4; ++t)
    bp[t] = W + (size_t)(h * 64 + t * 16 + lm) * ND + half * 16;

  v8f acc[2][4] = {};
  v16h a0 = mk16(*(const h8*)(arow0 + half * 16),
                 *(const h8*)(arow0 + 8 + half * 16));
  v16h a1 = mk16(*(const h8*)(arow1 + half * 16),
                 *(const h8*)(arow1 + 8 + half * 16));
  v16h bc[4];
#pragma unroll
  for (int t = 0; t < 4; ++t) bc[t] = *(const v16h*)(bp[t]);

  for (int kc = 0; kc < ND; kc += 32) {
    const int kn = (kc + 32 < ND) ? kc + 32 : kc;   // last iter: dummy reload
    __builtin_prefetch(arow0 + kn + 96, 0, 1);
    v16h a0n = mk16(*(const h8*)(arow0 + kn + half * 16),
                    *(const h8*)(arow0 + kn + 8 + half * 16));
    v16h a1n = mk16(*(const h8*)(arow1 + kn + half * 16),
                    *(const h8*)(arow1 + kn + 8 + half * 16));
    v16h bn[4];
#pragma unroll
    for (int t = 0; t < 4; ++t) bn[t] = *(const v16h*)(bp[t] + kn);
#pragma unroll
    for (int t = 0; t < 4; ++t) acc[0][t] = wmma16(a0, bc[t], acc[0][t]);
#pragma unroll
    for (int t = 0; t < 4; ++t) acc[1][t] = wmma16(a1, bc[t], acc[1][t]);
    a0 = a0n; a1 = a1n;
#pragma unroll
    for (int t = 0; t < 4; ++t) bc[t] = bn[t];
  }

#pragma unroll
  for (int i = 0; i < 2; ++i) {
#pragma unroll
    for (int t = 0; t < 4; ++t) {
      const int d   = t * 16 + lm;
      const int col = h * 64 + d;
      const float bb = bias[col];
#pragma unroll
      for (int r = 0; r < 8; ++r) {
        int m = m0 + i * 16 + r + half * 8;
        if (m < NM) {
          int b = m / NS;
          int s = m - b * NS;
          int bh = b * NH + h;
          float val = acc[i][t][r] + bb;
          if (proj == 0) {
            qo[((size_t)bh * NS + s) * NHD + d] = (_Float16)(val * 0.125f);
          } else if (proj == 1) {   // padded rows [NS,NSP) never written
            ko[((size_t)bh * NSP + s) * NHD + d] = (_Float16)val;
          } else {
            vo[((size_t)bh * NHD + d) * NSP + s] = (_Float16)val;
          }
        }
      }
    }
  }
}

// ---------------- stage 3: cls attention row (d_out tail) -------------------
__global__ void __launch_bounds__(256)
cls_row(const _Float16* __restrict__ qb, const _Float16* __restrict__ kb,
        float* __restrict__ cls) {
  __shared__ float scr[NS];
  __shared__ float red[256];
  __shared__ float q0s[NHD];
  const int bh  = blockIdx.x;
  const int tid = threadIdx.x;
  const _Float16* q0 = qb + (size_t)bh * NS * NHD;   // 1/8 already folded in
  const _Float16* kk = kb + (size_t)bh * NSP * NHD;
  if (tid < NHD) q0s[tid] = (float)q0[tid];
  __syncthreads();
  for (int j = tid; j < NS; j += 256) {
    const _Float16* kr = kk + (size_t)j * NHD;
    float acc = 0.f;
#pragma unroll
    for (int d = 0; d < NHD; ++d) acc += q0s[d] * (float)kr[d];
    scr[j] = acc;
  }
  __syncthreads();
  float mx = -1e30f;
  for (int j = tid; j < NS; j += 256) mx = fmaxf(mx, scr[j]);
  red[tid] = mx;
  __syncthreads();
  for (int s = 128; s > 0; s >>= 1) {
    if (tid < s) red[tid] = fmaxf(red[tid], red[tid + s]);
    __syncthreads();
  }
  mx = red[0];
  __syncthreads();
  float sm = 0.f;
  for (int j = tid; j < NS; j += 256) sm += __expf(scr[j] - mx);
  red[tid] = sm;
  __syncthreads();
  for (int s = 128; s > 0; s >>= 1) {
    if (tid < s) red[tid] += red[tid + s];
    __syncthreads();
  }
  float inv = 1.0f / red[0];
  for (int j = tid; j < NS - 1; j += 256)
    cls[(size_t)bh * 1024 + j] = __expf(scr[j + 1] - mx) * inv;
}

// ---------------- stage 4: attention core -----------------------------------
// grid: (65 q-tiles, H, B), 1 wave. K tiles staged into LDS by the Tensor
// Data Mover, double buffered with s_wait_tensorcnt<=1; PV V-fragments are
// register double-buffered.
__global__ void __launch_bounds__(32)
attn_core(const _Float16* __restrict__ qb, const _Float16* __restrict__ kb,
          const _Float16* __restrict__ vb, _Float16* __restrict__ ao) {
  __shared__ alignas(32)  _Float16 sc[16][NSP];     // 33 KB probability tile
  __shared__ alignas(128) _Float16 kl[2][16][NHD];  // 4 KB TDM K staging
  __shared__ float invs[16];
  const int qt   = blockIdx.x;
  const int h    = blockIdx.y;
  const int b    = blockIdx.z;
  const int bh   = b * NH + h;
  const int lane = threadIdx.x;
  const int half = lane >> 4;
  const int lm   = lane & 15;
  const _Float16* kbase = kb + (size_t)bh * NSP * NHD;

  // Q fragment (A-matrix, K = HD = 64 -> two v16h), rows clamped to S-1.
  int qr = qt * 16 + lm; if (qr > NS - 1) qr = NS - 1;
  const _Float16* qrow = qb + ((size_t)bh * NS + qr) * NHD;
  v16h aq0 = mk16(*(const h8*)(qrow + half * 16),
                  *(const h8*)(qrow + 8 + half * 16));
  v16h aq1 = mk16(*(const h8*)(qrow + 32 + half * 16),
                  *(const h8*)(qrow + 40 + half * 16));

  // ---- scores: 65 j-tiles of 16, K tile via TDM, double buffered ----
  tdm_load_k16x64(kbase, (unsigned)(size_t)&kl[0][0][0]);
  for (int jt = 0; jt < 65; ++jt) {
    const int slot = jt & 1;
    if (jt + 1 < 65) {
      tdm_load_k16x64(kbase + (size_t)(jt + 1) * 16 * NHD,
                      (unsigned)(size_t)&kl[slot ^ 1][0][0]);
      __builtin_amdgcn_s_wait_tensorcnt(1);   // current tile complete
    } else {
      __builtin_amdgcn_s_wait_tensorcnt(0);
    }
    asm volatile("" ::: "memory");            // keep LDS reads below the wait
    v16h b0 = *(const v16h*)&kl[slot][lm][half * 16];
    v16h b1 = *(const v16h*)&kl[slot][lm][half * 16 + 32];
    v8f s = {};
    s = wmma16(aq0, b0, s);
    s = wmma16(aq1, b1, s);
    const int j = jt * 16 + lm;
    if (j < NS) {                 // pad columns (garbage K rows) masked out
#pragma unroll
      for (int r = 0; r < 8; ++r) sc[r + half * 8][j] = (_Float16)s[r];
    }
  }
  __syncthreads();

  // ---- softmax per row; 1/sum deferred to epilogue ----
  if (lane < 16) {
    float mx = -1e30f;
    for (int j = 0; j < NS; ++j) mx = fmaxf(mx, (float)sc[lane][j]);
    float sum = 0.f;
    for (int j = 0; j < NS; ++j) {
      float e = __expf((float)sc[lane][j] - mx);
      sum += e;
      sc[lane][j] = (_Float16)e;
    }
    for (int j = NS; j < NSP; ++j) sc[lane][j] = (_Float16)0.f;
    invs[lane] = 1.0f / sum;
  }
  __syncthreads();

  // ---- PV: P[16 x NSP] @ V^T, 4 d-tiles, K chunks of 32, pipelined ----
  v8f od[4] = {};
  const _Float16* vp[4];
#pragma unroll
  for (int t = 0; t < 4; ++t)
    vp[t] = vb + ((size_t)bh * NHD + t * 16 + lm) * NSP + half * 16;
  v16h vc[4];
#pragma unroll
  for (int t = 0; t < 4; ++t) vc[t] = *(const v16h*)(vp[t]);
  for (int jc = 0; jc < NSP; jc += 32) {
    const int jn = (jc + 32 < NSP) ? jc + 32 : jc;
    v16h vn[4];
#pragma unroll
    for (int t = 0; t < 4; ++t) vn[t] = *(const v16h*)(vp[t] + jn);
    v16h pa = mk16(*(const h8*)&sc[lm][jc + half * 16],
                   *(const h8*)&sc[lm][jc + 8 + half * 16]);
#pragma unroll
    for (int t = 0; t < 4; ++t) od[t] = wmma16(pa, vc[t], od[t]);
#pragma unroll
    for (int t = 0; t < 4; ++t) vc[t] = vn[t];
  }

#pragma unroll
  for (int t = 0; t < 4; ++t) {
    const int d = t * 16 + lm;
#pragma unroll
    for (int r = 0; r < 8; ++r) {
      int srow = qt * 16 + r + half * 8;
      if (srow < NS) {
        float val = od[t][r] * invs[r + half * 8];
        ao[((size_t)b * NS + srow) * ND + h * NHD + d] = (_Float16)val;
      }
    }
  }
}

// ---------------- stage 5: output projection, fp32 result -------------------
__global__ void __launch_bounds__(32)
out_gemm(const _Float16* __restrict__ ah, const _Float16* __restrict__ wo,
         const float* __restrict__ bo, float* __restrict__ out) {
  const int h    = blockIdx.y;
  const int m0   = blockIdx.x * 32;
  const int lane = threadIdx.x;
  const int half = lane >> 4;
  const int lm   = lane & 15;

  int ra0 = m0 + lm;      if (ra0 > NM - 1) ra0 = NM - 1;
  int ra1 = m0 + 16 + lm; if (ra1 > NM - 1) ra1 = NM - 1;
  const _Float16* arow0 = ah + (size_t)ra0 * ND;
  const _Float16* arow1 = ah + (size_t)ra1 * ND;
  const _Float16* bp[4];
#pragma unroll
  for (int t = 0; t < 4; ++t)
    bp[t] = wo + (size_t)(h * 64 + t * 16 + lm) * ND + half * 16;

  v8f acc[2][4] = {};
  v16h a0 = mk16(*(const h8*)(arow0 + half * 16),
                 *(const h8*)(arow0 + 8 + half * 16));
  v16h a1 = mk16(*(const h8*)(arow1 + half * 16),
                 *(const h8*)(arow1 + 8 + half * 16));
  v16h bc[4];
#pragma unroll
  for (int t = 0; t < 4; ++t) bc[t] = *(const v16h*)(bp[t]);

  for (int kc = 0; kc < ND; kc += 32) {
    const int kn = (kc + 32 < ND) ? kc + 32 : kc;
    __builtin_prefetch(arow0 + kn + 96, 0, 1);
    v16h a0n = mk16(*(const h8*)(arow0 + kn + half * 16),
                    *(const h8*)(arow0 + kn + 8 + half * 16));
    v16h a1n = mk16(*(const h8*)(arow1 + kn + half * 16),
                    *(const h8*)(arow1 + kn + 8 + half * 16));
    v16h bn[4];
#pragma unroll
    for (int t = 0; t < 4; ++t) bn[t] = *(const v16h*)(bp[t] + kn);
#pragma unroll
    for (int t = 0; t < 4; ++t) acc[0][t] = wmma16(a0, bc[t], acc[0][t]);
#pragma unroll
    for (int t = 0; t < 4; ++t) acc[1][t] = wmma16(a1, bc[t], acc[1][t]);
    a0 = a0n; a1 = a1n;
#pragma unroll
    for (int t = 0; t < 4; ++t) bc[t] = bn[t];
  }

#pragma unroll
  for (int i = 0; i < 2; ++i) {
#pragma unroll
    for (int t = 0; t < 4; ++t) {
      const int col = h * 64 + t * 16 + lm;
      const float bb = bo[col];
#pragma unroll
      for (int r = 0; r < 8; ++r) {
        int m = m0 + i * 16 + r + half * 8;
        if (m < NM) out[(size_t)m * ND + col] = acc[i][t][r] + bb;
      }
    }
  }
}

// ---------------- launch ----------------------------------------------------
extern "C" void kernel_launch(void* const* d_in, const int* in_sizes, int n_in,
                              void* d_out, int out_size, void* d_ws,
                              size_t ws_size, hipStream_t stream) {
  (void)in_sizes; (void)n_in; (void)out_size; (void)ws_size;
  const float* x  = (const float*)d_in[0];
  const float* Wq = (const float*)d_in[1];
  const float* bq = (const float*)d_in[2];
  const float* Wk = (const float*)d_in[3];
  const float* bk = (const float*)d_in[4];
  const float* Wv = (const float*)d_in[5];
  const float* bv = (const float*)d_in[6];
  const float* Wo = (const float*)d_in[7];
  const float* bo = (const float*)d_in[8];

  float* out = (float*)d_out;
  float* cls = out + (size_t)NM * ND;

  constexpr size_t D2    = (size_t)ND * ND;
  constexpr size_t SZ_XH = (size_t)NM * ND * 2;
  constexpr size_t SZ_W  = 4 * D2 * 2;
  constexpr size_t SZ_Q  = (size_t)NB * NH * NS * NHD * 2;
  constexpr size_t SZ_KP = (size_t)NB * NH * NSP * NHD * 2;  // padded rows
  constexpr size_t SZ_V  = (size_t)NB * NH * NHD * NSP * 2;
  constexpr size_t OFF_XH = 0;
  constexpr size_t OFF_W  = OFF_XH + SZ_XH;
  constexpr size_t OFF_Q  = OFF_W + SZ_W;
  constexpr size_t OFF_K  = OFF_Q + SZ_Q;
  constexpr size_t OFF_V  = OFF_K + SZ_KP;
  constexpr size_t OFF_AO = OFF_V + SZ_V;

  char* ws = (char*)d_ws;
  _Float16* xh   = (_Float16*)(ws + OFF_XH);
  _Float16* wh   = (_Float16*)(ws + OFF_W);
  _Float16* qbuf = (_Float16*)(ws + OFF_Q);
  _Float16* kbuf = (_Float16*)(ws + OFF_K);
  _Float16* vbuf = (_Float16*)(ws + OFF_V);
  _Float16* abuf = (_Float16*)(ws + OFF_AO);

  // zero V^T so padded columns contribute exactly 0 in PV WMMAs
  hipMemsetAsync(vbuf, 0, SZ_V, stream);

  cvt_f32_f16<<<4096, 256, 0, stream>>>(x, xh, NM * ND);
  cvt_f32_f16<<<1024, 256, 0, stream>>>(Wq, wh + 0 * D2, (int)D2);
  cvt_f32_f16<<<1024, 256, 0, stream>>>(Wk, wh + 1 * D2, (int)D2);
  cvt_f32_f16<<<1024, 256, 0, stream>>>(Wv, wh + 2 * D2, (int)D2);
  cvt_f32_f16<<<1024, 256, 0, stream>>>(Wo, wh + 3 * D2, (int)D2);

  qkv_gemm<<<dim3((NM + 31) / 32, NH, 3), 32, 0, stream>>>(
      xh, wh, bq, bk, bv, qbuf, kbuf, vbuf);
  cls_row<<<NB * NH, 256, 0, stream>>>(qbuf, kbuf, cls);
  attn_core<<<dim3(65, NH, NB), 32, 0, stream>>>(qbuf, kbuf, vbuf, abuf);
  out_gemm<<<dim3((NM + 31) / 32, NH), 32, 0, stream>>>(abuf, wh + 3 * D2, bo,
                                                        out);
}
